// SimpleCRA_68994354643359
// MI455X (gfx1250) — matrix operand order repaired and checked by reading, not compiled, not run on hardware
//
#include <hip/hip_runtime.h>

typedef _Float16 half_t;
typedef __attribute__((ext_vector_type(16))) _Float16 v16h;
typedef __attribute__((ext_vector_type(8)))  _Float16 v8h;
typedef __attribute__((ext_vector_type(4)))  _Float16 v4h;
typedef __attribute__((ext_vector_type(8)))  float    v8f;
typedef __attribute__((ext_vector_type(4)))  float    v4f;

#define B_   16
#define L_   4096
#define D_   512
#define G_   (L_ / 2)        // 2048 word groups per sequence
#define M_   (B_ * G_)       // 32768 total word rows
#define K_   2048            // codebook entries

// d_out layout: [0, M_) indices (as float), [M_, M_ + M_*D_) word embeddings, last: loss
static const size_t OUT_EMB_OFF  = (size_t)M_;
static const size_t OUT_LOSS_OFF = (size_t)M_ + (size_t)M_ * D_;

// ---------------------------------------------------------------------------
// Prep 1: codebook -> f16, per-row squared norms, zero the loss accumulator.
// One wave per codebook row (8 waves / 256-thread block).
// ---------------------------------------------------------------------------
__global__ __launch_bounds__(256)
void cra_prep_cb(const float* __restrict__ cb, half_t* __restrict__ cbF16,
                 float* __restrict__ cb_sq, float* __restrict__ d_out) {
    if (blockIdx.x == 0 && threadIdx.x == 0) d_out[OUT_LOSS_OFF] = 0.0f;
    const int row  = (int)(blockIdx.x * (blockDim.x >> 5) + (threadIdx.x >> 5));
    const int lane = (int)(threadIdx.x & 31);
    if (row >= K_) return;
    const float* src = cb + (size_t)row * D_;
    half_t*      dst = cbF16 + (size_t)row * D_;
    float s = 0.0f;
#pragma unroll
    for (int j = 0; j < 4; ++j) {
        const int c = lane * 16 + j * 4;
        v4f v = *(const v4f*)(src + c);
        v4h h = { (half_t)v.x, (half_t)v.y, (half_t)v.z, (half_t)v.w };
        *(v4h*)(dst + c) = h;
        s += v.x * v.x + v.y * v.y + v.z * v.z + v.w * v.w;
    }
#pragma unroll
    for (int off = 16; off >= 1; off >>= 1) s += __shfl_xor(s, off, 32);
    if (lane == 0) cb_sq[row] = s;
}

// ---------------------------------------------------------------------------
// Prep 2: proj_w -> f16 (elementwise, vectorized).
// ---------------------------------------------------------------------------
__global__ __launch_bounds__(256)
void cra_prep_w(const float* __restrict__ w, half_t* __restrict__ wF16) {
    const int t = (int)(blockIdx.x * blockDim.x + threadIdx.x);
    if (t >= (D_ * D_) / 4) return;
    v4f v = *(const v4f*)(w + (size_t)t * 4);
    v4h h = { (half_t)v.x, (half_t)v.y, (half_t)v.z, (half_t)v.w };
    *(v4h*)(wF16 + (size_t)t * 4) = h;
}

// ---------------------------------------------------------------------------
// GEMM1 + argmin. Each wave owns TWO 16-row tiles of `means` (32 rows), so
// every B fragment (codebook tile) is reused by two independent WMMA chains:
// halves L2 codebook traffic and doubles WMMA ILP (no D->A/B hazards).
// means are computed on the fly (pair-mean of char_embeddings, non-temporal
// reads) into LDS as f16; argmin fused via score = ||c||^2 - 2*dot.
// Block = 2 waves, LDS = 2 * 32 * 512 * 2B = 64 KiB.
// ---------------------------------------------------------------------------
__global__ __launch_bounds__(64)
void cra_assign(const float* __restrict__ ce, const half_t* __restrict__ cbF16,
                const float* __restrict__ cb_sq, int* __restrict__ idx_ws,
                float* __restrict__ d_out) {
    __shared__ half_t lds[2 * 32 * D_];                   // 64 KiB / block
    const int lane   = (int)(threadIdx.x & 31);
    const int wave   = (int)(threadIdx.x >> 5);
    const int m0     = ((int)blockIdx.x * 2 + wave) * 32; // wave's first word row
    const int laneHi = lane >> 4;
    const int laneLo = lane & 15;
    half_t* A = lds + wave * 32 * D_;

    // Stage 32 means rows (f16) into LDS; char_embeddings read once, NT (use-once).
    for (int i = lane; i < (32 * D_) / 8; i += 32) {
        const int row = (i * 8) / D_;
        const int col = (i * 8) % D_;
        const float* p0 = ce + (size_t)(2 * (m0 + row)) * D_ + col;
        const float* p1 = p0 + D_;
        v4f a0 = __builtin_nontemporal_load((const v4f*)p0);
        v4f a1 = __builtin_nontemporal_load((const v4f*)(p0 + 4));
        v4f b0 = __builtin_nontemporal_load((const v4f*)p1);
        v4f b1 = __builtin_nontemporal_load((const v4f*)(p1 + 4));
        v8h h = {};
        h[0] = (half_t)(0.5f * (a0.x + b0.x));
        h[1] = (half_t)(0.5f * (a0.y + b0.y));
        h[2] = (half_t)(0.5f * (a0.z + b0.z));
        h[3] = (half_t)(0.5f * (a0.w + b0.w));
        h[4] = (half_t)(0.5f * (a1.x + b1.x));
        h[5] = (half_t)(0.5f * (a1.y + b1.y));
        h[6] = (half_t)(0.5f * (a1.z + b1.z));
        h[7] = (half_t)(0.5f * (a1.w + b1.w));
        *(v8h*)(A + i * 8) = h;
    }
    __syncthreads();

    float best[16];
    int   bidx[16];
#pragma unroll
    for (int r = 0; r < 16; ++r) { best[r] = 3.4e38f; bidx[r] = 0; }

    const half_t* Ap0 = A + laneLo * D_ + laneHi * 16;         // tile 0 rows 0..15
    const half_t* Ap1 = Ap0 + 16 * D_;                         // tile 1 rows 16..31

    for (int nt = 0; nt < K_ / 16; ++nt) {
        const int   ncol = nt * 16 + laneLo;                   // this lane's column
        const float csq  = cb_sq[ncol];
        const half_t* Bp = cbF16 + (size_t)ncol * D_ + laneHi * 16;
        v8f acc0 = {}, acc1 = {};
#pragma unroll 4
        for (int ks = 0; ks < D_ / 32; ++ks) {
            const int k = ks * 32;
            v16h b  = *(const v16h*)(Bp + k);                  // one B frag, used twice
            v8h a0l = *(const v8h*)(Ap0 + k);
            v8h a0h = *(const v8h*)(Ap0 + k + 8);
            v16h a0 = __builtin_shufflevector(a0l, a0h, 0,1,2,3,4,5,6,7,8,9,10,11,12,13,14,15);
            v8h a1l = *(const v8h*)(Ap1 + k);
            v8h a1h = *(const v8h*)(Ap1 + k + 8);
            v16h a1 = __builtin_shufflevector(a1l, a1h, 0,1,2,3,4,5,6,7,8,9,10,11,12,13,14,15);
            acc0 = __builtin_amdgcn_wmma_f32_16x16x32_f16(false, a0, false, b,
                                                          (short)0, acc0, false, false);
            acc1 = __builtin_amdgcn_wmma_f32_16x16x32_f16(false, a1, false, b,
                                                          (short)0, acc1, false, false);
        }
#pragma unroll
        for (int r = 0; r < 8; ++r) {
            const float s0 = csq - 2.0f * acc0[r];
            if (s0 < best[r])     { best[r]     = s0; bidx[r]     = ncol; }
            const float s1 = csq - 2.0f * acc1[r];
            if (s1 < best[8 + r]) { best[8 + r] = s1; bidx[8 + r] = ncol; }
        }
    }

    // argmin across the 16 lanes of each half-wave (ties -> smaller index)
#pragma unroll
    for (int r = 0; r < 16; ++r) {
#pragma unroll
        for (int off = 8; off >= 1; off >>= 1) {
            const float ov = __shfl_xor(best[r], off, 32);
            const int   oi = __shfl_xor(bidx[r], off, 32);
            if (ov < best[r] || (ov == best[r] && oi < bidx[r])) {
                best[r] = ov; bidx[r] = oi;
            }
        }
    }
    if (laneLo == 0) {
#pragma unroll
        for (int r = 0; r < 8; ++r) {
            const int ma = m0 + laneHi * 8 + r;                // tile 0
            idx_ws[ma] = bidx[r];
            d_out[ma]  = (float)bidx[r];
            const int mb = m0 + 16 + laneHi * 8 + r;           // tile 1
            idx_ws[mb] = bidx[8 + r];
            d_out[mb]  = (float)bidx[8 + r];
        }
    }
}

// ---------------------------------------------------------------------------
// Gather + GEMM2 + MSE. Each wave owns TWO 16-row tiles (32 rows): one
// proj_w fragment feeds two WMMA chains. word_embeddings streamed to d_out
// with non-temporal stores (write-once, keep L2 for codebook/weights).
// ---------------------------------------------------------------------------
__global__ __launch_bounds__(128)
void cra_project(const float* __restrict__ cb, const half_t* __restrict__ cbF16,
                 const half_t* __restrict__ wF16, const float* __restrict__ bias,
                 const int* __restrict__ idx_ws, float* __restrict__ d_out) {
    const int lane   = (int)(threadIdx.x & 31);
    const int wave   = (int)(threadIdx.x >> 5);
    const int m0     = ((int)blockIdx.x * 4 + wave) * 32;
    const int laneHi = lane >> 4;
    const int laneLo = lane & 15;

    // Stream word_embeddings (f32 gather from codebook) to output, NT stores.
    float* outEmb = d_out + OUT_EMB_OFF;
    for (int i = lane; i < (32 * D_) / 4; i += 32) {
        const int row = (i * 4) / D_;
        const int col = (i * 4) % D_;
        const int ci  = idx_ws[m0 + row];
        v4f v = *(const v4f*)(cb + (size_t)ci * D_ + col);
        __builtin_nontemporal_store(v, (v4f*)(outEmb + (size_t)(m0 + row) * D_ + col));
    }

    const int myIdx0 = idx_ws[m0 + laneLo];                    // tile 0 A-row gather
    const int myIdx1 = idx_ws[m0 + 16 + laneLo];               // tile 1 A-row gather
    int rowIdx[16];
#pragma unroll
    for (int t = 0; t < 2; ++t)
#pragma unroll
        for (int r = 0; r < 8; ++r)
            rowIdx[t * 8 + r] = idx_ws[m0 + t * 16 + laneHi * 8 + r];

    const half_t* Ap0 = cbF16 + (size_t)myIdx0 * D_ + laneHi * 16;
    const half_t* Ap1 = cbF16 + (size_t)myIdx1 * D_ + laneHi * 16;
    float lsum = 0.0f;

    for (int nt = 0; nt < D_ / 16; ++nt) {
        const int ecol   = nt * 16 + laneLo;
        const half_t* Bp = wF16 + (size_t)ecol * D_ + laneHi * 16;
        v8f acc0 = {}, acc1 = {};
#pragma unroll 4
        for (int ks = 0; ks < D_ / 32; ++ks) {
            const int k = ks * 32;
            v16h b  = *(const v16h*)(Bp + k);                  // one B frag, used twice
            v8h a0l = *(const v8h*)(Ap0 + k);
            v8h a0h = *(const v8h*)(Ap0 + k + 8);
            v16h a0 = __builtin_shufflevector(a0l, a0h, 0,1,2,3,4,5,6,7,8,9,10,11,12,13,14,15);
            v8h a1l = *(const v8h*)(Ap1 + k);
            v8h a1h = *(const v8h*)(Ap1 + k + 8);
            v16h a1 = __builtin_shufflevector(a1l, a1h, 0,1,2,3,4,5,6,7,8,9,10,11,12,13,14,15);
            acc0 = __builtin_amdgcn_wmma_f32_16x16x32_f16(false, a0, false, b,
                                                          (short)0, acc0, false, false);
            acc1 = __builtin_amdgcn_wmma_f32_16x16x32_f16(false, a1, false, b,
                                                          (short)0, acc1, false, false);
        }
        const float be = bias[ecol];
#pragma unroll
        for (int r = 0; r < 8; ++r) {
            const float t0 = cb[(size_t)rowIdx[r] * D_ + ecol];
            const float d0 = acc0[r] + be - t0;
            lsum += d0 * d0;
            const float t1 = cb[(size_t)rowIdx[8 + r] * D_ + ecol];
            const float d1 = acc1[r] + be - t1;
            lsum += d1 * d1;
        }
    }
#pragma unroll
    for (int off = 16; off >= 1; off >>= 1) lsum += __shfl_xor(lsum, off, 32);
    if (lane == 0)
        atomicAdd(d_out + OUT_LOSS_OFF, lsum * (1.0f / ((float)M_ * (float)D_)));
}

// ---------------------------------------------------------------------------
extern "C" void kernel_launch(void* const* d_in, const int* in_sizes, int n_in,
                              void* d_out, int out_size, void* d_ws, size_t ws_size,
                              hipStream_t stream) {
    (void)in_sizes; (void)n_in; (void)out_size; (void)ws_size;
    // inputs: [0]=char_tokens (unused), [1]=char_embeddings, [2]=word_codebook,
    //         [3]=proj_w, [4]=proj_b
    const float* char_emb = (const float*)d_in[1];
    const float* codebook = (const float*)d_in[2];
    const float* proj_w   = (const float*)d_in[3];
    const float* proj_b   = (const float*)d_in[4];
    float*       out      = (float*)d_out;

    // workspace carve-up (~2.7 MiB total)
    uint8_t* ws    = (uint8_t*)d_ws;
    half_t*  cbF16 = (half_t*)ws;                              // 2 MiB
    half_t*  wF16  = (half_t*)(ws + (size_t)K_ * D_ * 2);      // 512 KiB
    float*   cb_sq = (float*)(ws + (size_t)K_ * D_ * 2 + (size_t)D_ * D_ * 2);
    int*     idx   = (int*)((uint8_t*)cb_sq + (size_t)K_ * 4);

    cra_prep_cb<<<K_ / 8, 256, 0, stream>>>(codebook, cbF16, cb_sq, out);
    cra_prep_w<<<(D_ * D_ / 4) / 256, 256, 0, stream>>>(proj_w, wF16);
    // 2 waves/block, 32 rows/wave  -> 512 blocks
    cra_assign<<<M_ / 64, 64, 0, stream>>>(char_emb, cbF16, cb_sq, idx, out);
    // 4 waves/block, 32 rows/wave  -> 256 blocks
    cra_project<<<M_ / 128, 128, 0, stream>>>(codebook, cbF16, wF16, proj_b, idx, out);
}
